// SemiBilinearModular_68805376081923
// MI455X (gfx1250) — matrix operand order repaired and checked by reading, not compiled, not run on hardware
//
#include <hip/hip_runtime.h>

typedef __attribute__((ext_vector_type(16))) _Float16 v16h;
typedef __attribute__((ext_vector_type(8)))  _Float16 v8h;
typedef __attribute__((ext_vector_type(8)))  float    v8f;

#define NSTEP  256
#define H1     64
#define Z1     256    // 4*H1
#define H2     256
#define Z2     1024   // 4*H2
#define EMB    20

// B-fragment (32x16 f16) for v_wmma_f32_16x16x32_f16:
// lane L holds column n = L%16; lanes 0-15: K {k0+0..7, k0+16..23}; lanes 16-31: K {k0+8..15, k0+24..31}.
__device__ __forceinline__ v16h load_bfrag(const float* __restrict__ W, int K, int ldw,
                                           int k0, int col, int kbase) {
  v16h f;
#pragma unroll
  for (int j = 0; j < 16; ++j) {
    int k = k0 + (j < 8 ? (kbase >> 1) + j : 8 + (kbase >> 1) + j);
    float v = (k < K) ? W[k * ldw + col] : 0.0f;
    f[j] = (_Float16)v;
  }
  return f;
}

// A-fragment (16x32 f16): lane L holds row m = L%16;
// lanes 0-15: K {k0+0..7, k0+16..23}; lanes 16-31: K {k0+8..15, k0+24..31}.
// Two 16-byte LDS loads per fragment.
__device__ __forceinline__ v16h load_afrag(const _Float16* row, int k0, int kboff) {
  const v8h lo = *(const v8h*)(row + k0 + kboff);
  const v8h hi = *(const v8h*)(row + k0 + 16 + kboff);
  return __builtin_shufflevector(lo, hi, 0, 1, 2, 3, 4, 5, 6, 7,
                                         8, 9, 10, 11, 12, 13, 14, 15);
}

__global__ __launch_bounds__(256) void sbm_fused(
    const float* __restrict__ x,
    const float* __restrict__ We_a, const float* __restrict__ We_b,
    const float* __restrict__ Wx_a64, const float* __restrict__ Wh_a64, const float* __restrict__ b_a64,
    const float* __restrict__ Wx_b64, const float* __restrict__ Wh_b64, const float* __restrict__ b_b64,
    const float* __restrict__ Wx_a256, const float* __restrict__ b_a256,
    const float* __restrict__ Wx_b256, const float* __restrict__ b_b256,
    const float* __restrict__ W_out, const float* __restrict__ b_out,
    float* __restrict__ out) {
  __shared__ __align__(16) float     xb[16][512];     // this block's x rows (loaded once)
  __shared__ __align__(16) float     zs[16][Z1];      // z = za*zb staging
  __shared__ __align__(16) _Float16  hbuf[16][H1];    // h state as f16 (A-operand source)
  __shared__ __align__(16) _Float16  xf[16][32];      // embedded x_t fragment (K padded 20->32)
  __shared__ float sWe_a[2][EMB], sWe_b[2][EMB];

  const int tid   = threadIdx.x;
  const int lane  = tid & 31;
  const int w     = tid >> 5;       // wave id 0..7
  const int n     = lane & 15;      // fragment column / row index
  const int hi16  = lane >> 4;
  const int mb    = hi16 ? 8 : 0;   // C/D row base
  const int kboff = hi16 ? 8 : 0;   // A-frag K interleave offset
  const int kbase = hi16 ? 16 : 0;  // B-frag K half offset
  const int m_a   = n;              // A-frag row
  const int r0    = blockIdx.x * 16;
  const int n0    = w * 32;         // this wave's column base in z

  // ---- time-invariant B-fragments + biases, resident in VGPRs for all 256 steps ----
  v16h WxA[2], WxB[2], WhA0[2], WhA1[2], WhB0[2], WhB1[2];
  float bA[2], bB[2];
#pragma unroll
  for (int i = 0; i < 2; ++i) {
    const int col = n0 + 16 * i + n;
    WxA[i]  = load_bfrag(Wx_a64, EMB, Z1, 0,  col, kbase);
    WxB[i]  = load_bfrag(Wx_b64, EMB, Z1, 0,  col, kbase);
    WhA0[i] = load_bfrag(Wh_a64, H1,  Z1, 0,  col, kbase);
    WhA1[i] = load_bfrag(Wh_a64, H1,  Z1, 32, col, kbase);
    WhB0[i] = load_bfrag(Wh_b64, H1,  Z1, 0,  col, kbase);
    WhB1[i] = load_bfrag(Wh_b64, H1,  Z1, 32, col, kbase);
    bA[i] = b_a64[col];
    bB[i] = b_b64[col];
  }

  // ---- one-time LDS setup ----
  if (tid < 2 * EMB) {
    sWe_a[tid / EMB][tid % EMB] = We_a[tid];
    sWe_b[tid / EMB][tid % EMB] = We_b[tid];
  }
  for (int idx = tid; idx < 16 * H1; idx += 256)
    (&hbuf[0][0])[idx] = (_Float16)0.0f;
  for (int idx = tid; idx < 16 * 128; idx += 256) {  // 16 rows x 512 floats as float4
    const int m = idx >> 7, c4 = idx & 127;
    ((float4*)&xb[m][0])[c4] = ((const float4*)(x + (size_t)(r0 + m) * 512))[c4];
  }

  // Warm L2 with stage-2 weights while the recurrence runs (gfx1250 prefetch path).
  {
    const size_t off = ((size_t)(blockIdx.x * 256 + tid) * 128) & (size_t)(H1 * Z2 * 4 - 1);
    __builtin_prefetch((const char*)Wx_a256 + off, 0, 1);
    __builtin_prefetch((const char*)Wx_b256 + off, 0, 1);
  }
  __syncthreads();

  // embedded x fragment for t = 0
#pragma unroll
  for (int it = 0; it < 2; ++it) {
    const int idx = tid + 256 * it;
    const int m = idx >> 5, e = idx & 31;
    float v = 0.0f;
    if (e < EMB) {
      const float x0 = xb[m][0], x1 = xb[m][1];
      v = (x0 * sWe_a[0][e] + x1 * sWe_a[1][e]) * (x0 * sWe_b[0][e] + x1 * sWe_b[1][e]);
    }
    xf[m][e] = (_Float16)v;
  }
  __syncthreads();

  // per-thread c state: row m_g, 4 consecutive columns starting at j0
  const int m_g = tid >> 4;
  const int j0  = (tid & 15) * 4;
  float cst[4] = {0.f, 0.f, 0.f, 0.f};

  // ======================= 256-step recurrence =======================
  for (int t = 0; t < NSTEP; ++t) {
    const v16h xa  = load_afrag(&xf[m_a][0],   0,  kboff);
    const v16h ha0 = load_afrag(&hbuf[m_a][0], 0,  kboff);
    const v16h ha1 = load_afrag(&hbuf[m_a][0], 32, kboff);

#pragma unroll
    for (int i = 0; i < 2; ++i) {
      v8f acca = {};
      v8f accb = {};
      acca = __builtin_amdgcn_wmma_f32_16x16x32_f16(false, xa,  false, WxA[i],  (short)0, acca, false, false);
      acca = __builtin_amdgcn_wmma_f32_16x16x32_f16(false, ha0, false, WhA0[i], (short)0, acca, false, false);
      acca = __builtin_amdgcn_wmma_f32_16x16x32_f16(false, ha1, false, WhA1[i], (short)0, acca, false, false);
      accb = __builtin_amdgcn_wmma_f32_16x16x32_f16(false, xa,  false, WxB[i],  (short)0, accb, false, false);
      accb = __builtin_amdgcn_wmma_f32_16x16x32_f16(false, ha0, false, WhB0[i], (short)0, accb, false, false);
      accb = __builtin_amdgcn_wmma_f32_16x16x32_f16(false, ha1, false, WhB1[i], (short)0, accb, false, false);
      const int ncol = n0 + 16 * i + n;
#pragma unroll
      for (int r = 0; r < 8; ++r)
        zs[mb + r][ncol] = (acca[r] + bA[i]) * (accb[r] + bB[i]);
    }
    __syncthreads();   // z ready; fragment reads done

    // gates: c = f*c + i*g ; h = o*c  (z slices: i|f|g|o of width 64)
#pragma unroll
    for (int kk = 0; kk < 4; ++kk) {
      const int j = j0 + kk;
      const float iv = zs[m_g][j];
      const float fv = zs[m_g][64 + j];
      const float gv = zs[m_g][128 + j];
      const float ov = zs[m_g][192 + j];
      cst[kk] = fv * cst[kk] + iv * gv;
      hbuf[m_g][j] = (_Float16)(ov * cst[kk]);
    }
    // embedded x fragment for t+1
    if (t + 1 < NSTEP) {
#pragma unroll
      for (int it = 0; it < 2; ++it) {
        const int idx = tid + 256 * it;
        const int m = idx >> 5, e = idx & 31;
        float v = 0.0f;
        if (e < EMB) {
          const float x0 = xb[m][2 * (t + 1)], x1 = xb[m][2 * (t + 1) + 1];
          v = (x0 * sWe_a[0][e] + x1 * sWe_a[1][e]) * (x0 * sWe_b[0][e] + x1 * sWe_b[1][e]);
        }
        xf[m][e] = (_Float16)v;
      }
    }
    __syncthreads();   // h/xf ready for next step
  }

  // ======================= stage 2: H2=256 bilinear LSTM step (h0=c0=0) =======================
  // z2 = (h64@Wx_a256 + b_a256) * (h64@Wx_b256 + b_b256); c2 = i*g; h2 = o*c2
  {
    const int m2 = tid >> 4, jb = tid & 15;
    float hr[H1];
#pragma unroll
    for (int k = 0; k < H1; ++k) hr[k] = (float)hbuf[m2][k];

    for (int s = 0; s < 16; ++s) {
      const int j = jb + 16 * s;
      float ai = b_a256[j],       bi = b_b256[j];
      float ag = b_a256[512 + j], bg = b_b256[512 + j];
      float ao = b_a256[768 + j], bo = b_b256[768 + j];
#pragma unroll 16
      for (int k = 0; k < H1; ++k) {
        const float h = hr[k];
        const float* ra = Wx_a256 + (size_t)k * Z2;
        const float* rb = Wx_b256 + (size_t)k * Z2;
        ai += h * ra[j];       bi += h * rb[j];
        ag += h * ra[512 + j]; bg += h * rb[512 + j];
        ao += h * ra[768 + j]; bo += h * rb[768 + j];
      }
      const float zi = ai * bi, zg = ag * bg, zo = ao * bo;
      zs[m2][j] = zo * (zi * zg);   // h2
    }
  }
  __syncthreads();

  // out = h2 @ W_out + b_out   (256 -> 2)
  if (tid < 32) {
    const int m = tid >> 1, kk = tid & 1;
    float s = b_out[kk];
#pragma unroll 16
    for (int j = 0; j < H2; ++j) s += zs[m][j] * W_out[j * 2 + kk];
    out[(size_t)(r0 + m) * 2 + kk] = s;
  }
}

extern "C" void kernel_launch(void* const* d_in, const int* in_sizes, int n_in,
                              void* d_out, int out_size, void* d_ws, size_t ws_size,
                              hipStream_t stream) {
  (void)in_sizes; (void)n_in; (void)out_size; (void)d_ws; (void)ws_size;
  const float* x       = (const float*)d_in[0];
  const float* We_a    = (const float*)d_in[1];
  const float* We_b    = (const float*)d_in[2];
  const float* Wx_a64  = (const float*)d_in[3];
  const float* Wh_a64  = (const float*)d_in[4];
  const float* b_a64   = (const float*)d_in[5];
  const float* Wx_b64  = (const float*)d_in[6];
  const float* Wh_b64  = (const float*)d_in[7];
  const float* b_b64   = (const float*)d_in[8];
  const float* Wx_a256 = (const float*)d_in[9];
  // d_in[10] Wh_a256: unused (h0_2 == 0)
  const float* b_a256  = (const float*)d_in[11];
  const float* Wx_b256 = (const float*)d_in[12];
  // d_in[13] Wh_b256: unused (h0_2 == 0)
  const float* b_b256  = (const float*)d_in[14];
  const float* W_out   = (const float*)d_in[15];
  const float* b_out   = (const float*)d_in[16];
  float* out = (float*)d_out;

  dim3 grid(128), block(256);
  sbm_fused<<<grid, block, 0, stream>>>(x, We_a, We_b,
                                        Wx_a64, Wh_a64, b_a64,
                                        Wx_b64, Wh_b64, b_b64,
                                        Wx_a256, b_a256,
                                        Wx_b256, b_b256,
                                        W_out, b_out, out);
}
